// InteractiveAttentionLayer_6296422056376
// MI455X (gfx1250) — matrix operand order repaired and checked by reading, not compiled, not run on hardware
//
#include <hip/hip_runtime.h>

typedef __bf16 bf16_t;
typedef __attribute__((ext_vector_type(16))) __bf16 v16bf;
typedef __attribute__((ext_vector_type(8)))  float  v8f;

constexpr int kS  = 4096;   // sequence length
constexpr int kD  = 256;    // head dim
constexpr int kB  = 4;      // batch
constexpr int kBR = 128;    // rows per workgroup (8 waves x 16)
constexpr int kBC = 32;     // kv columns per iteration
constexpr int kKSTR = kD + 8;    // padded K-tile row stride (elements)
constexpr int kVSTR = kBC + 8;   // padded Vt-tile row stride
constexpr int kPSTR = kBC + 8;   // padded P scratch row stride

static __device__ __forceinline__ v8f zero8() {
  v8f r;
#pragma unroll
  for (int i = 0; i < 8; ++i) r[i] = 0.0f;
  return r;
}

// Build a v16bf fragment from two 16-byte runs in LDS (each 16B aligned).
static __device__ __forceinline__ v16bf ld_frag(const bf16_t* p0, const bf16_t* p1) {
  union { v16bf v; uint4 q[2]; } u;
  u.q[0] = *(const uint4*)p0;
  u.q[1] = *(const uint4*)p1;
  return u.v;
}

static __device__ __forceinline__ v8f wmma_bf(v16bf a, v16bf b, v8f c) {
  // (neg_a, A, neg_b, B, c_mod, C, reuse_a, reuse_b)
  return __builtin_amdgcn_wmma_f32_16x16x32_bf16(false, a, false, b, (short)0, c,
                                                 false, false);
}

__global__ __launch_bounds__(256) void ia_init(const float* __restrict__ hb,
                                               const float* __restrict__ hp,
                                               float* __restrict__ out, int n) {
  int i = blockIdx.x * blockDim.x + threadIdx.x;
  if (i < n) out[i] = hb[i] + hp[i];
}

__global__ __launch_bounds__(256) void ia_flash(const float* __restrict__ hb,
                                                const float* __restrict__ hp,
                                                float* __restrict__ out) {
  // K tile (row-major, hi/lo bf16 split), V tile (transposed, hi/lo split)
  __shared__ bf16_t Khi[kBC * kKSTR], Klo[kBC * kKSTR];
  __shared__ bf16_t Vthi[kD * kVSTR], Vtlo[kD * kVSTR];
  // per-wave P scratch for C-layout -> A-layout conversion
  __shared__ bf16_t Pshi[8 * 16 * kPSTR], Pslo[8 * 16 * kPSTR];

  const int b   = blockIdx.y;
  const int dir = blockIdx.z;        // 0: Q=K=h_p,V=h_b   1: Q=K=h_b,V=h_p
  const int r0  = blockIdx.x * kBR;
  const float* Qp = dir ? hb : hp;
  const float* Vp = dir ? hp : hb;

  const int tid  = threadIdx.x;
  const int wave = tid >> 5;
  const int lane = tid & 31;
  const int lm   = lane & 15;        // M (A rows) or N (B/C cols) within tile
  const int lh   = lane >> 4;        // half-wave selector

  // staging assignment (also used for prefetch addresses)
  const int kv    = tid >> 3;        // 0..31 kv row
  const int dbase = (tid & 7) * 32;  // 8 threads/row, 32 d (=128B line) each
  const float* kbase_g = Qp + ((size_t)b * kS + kv) * (size_t)kD + dbase;
  const float* vbase_g = Vp + ((size_t)b * kS + kv) * (size_t)kD + dbase;

  // ---- Q rows -> registers in WMMA A-layout, fp32 split into bf16 hi+lo
  v16bf qhi[8], qlo[8];
  {
    const int row = r0 + wave * 16 + lm;
    const float* qr = Qp + ((size_t)b * kS + row) * (size_t)kD;
#pragma unroll
    for (int kc = 0; kc < 8; ++kc) {
      const int base = kc * 32 + lh * 8;
#pragma unroll
      for (int e = 0; e < 8; ++e) {
        float x0 = qr[base + e];
        float x1 = qr[base + 16 + e];
        bf16_t h0 = (bf16_t)x0, h1 = (bf16_t)x1;
        qhi[kc][e]     = h0;
        qhi[kc][e + 8] = h1;
        qlo[kc][e]     = (bf16_t)(x0 - (float)h0);
        qlo[kc][e + 8] = (bf16_t)(x1 - (float)h1);
      }
    }
  }

  v8f O[16];
#pragma unroll
  for (int t = 0; t < 16; ++t) O[t] = zero8();
  float mrow[8], lrow[8];
#pragma unroll
  for (int v = 0; v < 8; ++v) { mrow[v] = -1.0e30f; lrow[v] = 0.0f; }

  bf16_t* Pwh = Pshi + wave * (16 * kPSTR);
  bf16_t* Pwl = Pslo + wave * (16 * kPSTR);
  const float LOG2E = 1.4426950408889634f;

  for (int jc = 0; jc < kS; jc += kBC) {
    // ---- cooperative K/V tile stage: convert + hi/lo split (+ transpose for V)
    {
      const float4* kr = (const float4*)(kbase_g + (size_t)jc * kD);
      const float4* vr = (const float4*)(vbase_g + (size_t)jc * kD);
#pragma unroll
      for (int q4i = 0; q4i < 8; ++q4i) {
        float4 kx = kr[q4i];
        float4 vx = vr[q4i];
        float kk[4] = {kx.x, kx.y, kx.z, kx.w};
        float vv[4] = {vx.x, vx.y, vx.z, vx.w};
#pragma unroll
        for (int j = 0; j < 4; ++j) {
          int d = dbase + q4i * 4 + j;
          bf16_t kh = (bf16_t)kk[j];
          Khi[kv * kKSTR + d] = kh;
          Klo[kv * kKSTR + d] = (bf16_t)(kk[j] - (float)kh);
          bf16_t vh = (bf16_t)vv[j];
          Vthi[d * kVSTR + kv] = vh;
          Vtlo[d * kVSTR + kv] = (bf16_t)(vv[j] - (float)vh);
        }
      }
    }
    __syncthreads();

    // ---- prefetch next tile (global_prefetch_b8) while this tile computes;
    // uniform guard keeps EXEC all-ones for the WMMAs below.
    if (jc + kBC < kS) {
      __builtin_prefetch(kbase_g + (size_t)(jc + kBC) * kD, 0, 1);
      __builtin_prefetch(vbase_g + (size_t)(jc + kBC) * kD, 0, 1);
    }

    // ---- scores S = Q K^T  (fp32-accurate: Qh*Kh + Qh*Kl + Ql*Kh)
    v8f s0 = zero8(), s1 = zero8();
#pragma unroll
    for (int kc = 0; kc < 8; ++kc) {
      const int koff = kc * 32 + lh * 8;
      const bf16_t* k0h = Khi + lm * kKSTR + koff;
      const bf16_t* k0l = Klo + lm * kKSTR + koff;
      const bf16_t* k1h = Khi + (16 + lm) * kKSTR + koff;
      const bf16_t* k1l = Klo + (16 + lm) * kKSTR + koff;
      v16bf bh0 = ld_frag(k0h, k0h + 16);
      v16bf bl0 = ld_frag(k0l, k0l + 16);
      v16bf bh1 = ld_frag(k1h, k1h + 16);
      v16bf bl1 = ld_frag(k1l, k1l + 16);
      s0 = wmma_bf(qhi[kc], bh0, s0);
      s0 = wmma_bf(qhi[kc], bl0, s0);
      s0 = wmma_bf(qlo[kc], bh0, s0);
      s1 = wmma_bf(qhi[kc], bh1, s1);
      s1 = wmma_bf(qhi[kc], bl1, s1);
      s1 = wmma_bf(qlo[kc], bh1, s1);
    }

    // ---- online softmax over the 32 new columns (rows map: v + 8*lh)
    float scale[8];
#pragma unroll
    for (int v = 0; v < 8; ++v) {
      float mx = fmaxf(s0[v], s1[v]);
      mx = fmaxf(mx, __shfl_xor(mx, 1));
      mx = fmaxf(mx, __shfl_xor(mx, 2));
      mx = fmaxf(mx, __shfl_xor(mx, 4));
      mx = fmaxf(mx, __shfl_xor(mx, 8));
      float mn = fmaxf(mrow[v], mx);
      float sc = exp2f((mrow[v] - mn) * LOG2E);
      float p0 = exp2f((s0[v] - mn) * LOG2E);
      float p1 = exp2f((s1[v] - mn) * LOG2E);
      float rs = p0 + p1;
      rs += __shfl_xor(rs, 1);
      rs += __shfl_xor(rs, 2);
      rs += __shfl_xor(rs, 4);
      rs += __shfl_xor(rs, 8);
      lrow[v] = lrow[v] * sc + rs;
      mrow[v] = mn;
      scale[v] = sc;
      // C-layout -> LDS (P is 16x32, row-major, hi/lo split)
      const int prow = v + 8 * lh;
      bf16_t p0h = (bf16_t)p0, p1h = (bf16_t)p1;
      Pwh[prow * kPSTR + lm]      = p0h;
      Pwh[prow * kPSTR + 16 + lm] = p1h;
      Pwl[prow * kPSTR + lm]      = (bf16_t)(p0 - (float)p0h);
      Pwl[prow * kPSTR + 16 + lm] = (bf16_t)(p1 - (float)p1h);
    }
#pragma unroll
    for (int t = 0; t < 16; ++t)
#pragma unroll
      for (int v = 0; v < 8; ++v) O[t][v] *= scale[v];

    // ---- O += P V  (Ph*Vh + Ph*Vl + Pl*Vh); DS ops in-order within wave
    const bf16_t* pa0 = Pwh + lm * kPSTR + lh * 8;
    const bf16_t* pl0 = Pwl + lm * kPSTR + lh * 8;
    v16bf pah = ld_frag(pa0, pa0 + 16);
    v16bf pal = ld_frag(pl0, pl0 + 16);
#pragma unroll
    for (int t = 0; t < 16; ++t) {
      const bf16_t* vbh = Vthi + (t * 16 + lm) * kVSTR + lh * 8;
      const bf16_t* vbl = Vtlo + (t * 16 + lm) * kVSTR + lh * 8;
      v16bf vh = ld_frag(vbh, vbh + 16);
      v16bf vl = ld_frag(vbl, vbl + 16);
      O[t] = wmma_bf(pah, vh, O[t]);
      O[t] = wmma_bf(pah, vl, O[t]);
      O[t] = wmma_bf(pal, vh, O[t]);
    }
    __syncthreads();
  }

  // ---- normalize and accumulate (two directions share out -> atomic add)
#pragma unroll
  for (int v = 0; v < 8; ++v) {
    const float inv = 1.0f / lrow[v];
    const int row = r0 + wave * 16 + v + 8 * lh;
    float* orow = out + ((size_t)b * kS + row) * (size_t)kD;
#pragma unroll
    for (int t = 0; t < 16; ++t)
      atomicAdd(orow + t * 16 + lm, O[t][v] * inv);
  }
}

extern "C" void kernel_launch(void* const* d_in, const int* in_sizes, int n_in,
                              void* d_out, int out_size, void* d_ws, size_t ws_size,
                              hipStream_t stream) {
  (void)in_sizes; (void)n_in; (void)d_ws; (void)ws_size; (void)out_size;
  const float* hb = (const float*)d_in[0];
  const float* hp = (const float*)d_in[1];
  float* out = (float*)d_out;

  const int n = kB * kS * kD;
  ia_init<<<dim3((n + 255) / 256), dim3(256), 0, stream>>>(hb, hp, out, n);
  ia_flash<<<dim3(kS / kBR, kB, 2), dim3(256), 0, stream>>>(hb, hp, out);
}
